// TFBlock_2817498546609
// MI455X (gfx1250) — matrix operand order repaired and checked by reading, not compiled, not run on hardware
//
#include <hip/hip_runtime.h>
#include <hip/hip_bf16.h>
#include <stdint.h>

#define B_   2
#define S_   2048
#define D_   1024
#define H_   16
#define DH_  64
#define E_   8
#define F_   4096
#define NTOK (B_ * S_)

typedef __attribute__((ext_vector_type(16))) __bf16 v16bf_t;
typedef __attribute__((ext_vector_type(8)))  float  v8f_t;

union FragAB { v16bf_t v; unsigned int u[8]; unsigned short h[16]; };
union FragC  { v8f_t v; float f[8]; };

__device__ __forceinline__ unsigned short f2bf(float f) {
    unsigned int u = __float_as_uint(f);
    u += 0x7FFFu + ((u >> 16) & 1u);   // round-to-nearest-even
    return (unsigned short)(u >> 16);
}
__device__ __forceinline__ unsigned int pack2bf(float a, float b) {
    return (unsigned int)f2bf(a) | ((unsigned int)f2bf(b) << 16);
}

__device__ __forceinline__ float gelu_tanh(float x) {
    const float c0 = 0.7978845608028654f, c1 = 0.044715f;
    float x3 = x * x * x;
    return 0.5f * x * (1.0f + tanhf(c0 * (x + c1 * x3)));
}

__device__ __forceinline__ void wmma_bf16(FragC& c, const FragAB& a, const FragAB& b) {
    c.v = __builtin_amdgcn_wmma_f32_16x16x32_bf16(false, a.v, false, b.v,
                                                  (short)0, c.v, false, false);
}

// A-fragment K offset for vgpr v (16-bit A, 16x32): contiguous k = 16*hi + 2*v
__device__ __forceinline__ int akk(int v, int hi) {
    return (hi << 4) + (v << 1);
}

// ---------------------------------------------------------------------------
// f32 [K,N] -> pair-packed bf16 rows: dst[(k/2)*N + n] = {bf16(k,n), bf16(k+1,n)}
// This makes every WMMA B-fragment dword a single aligned b32 load.
// ---------------------------------------------------------------------------
__global__ void cvt_pack_rows(const float* __restrict__ src,
                              unsigned int* __restrict__ dst, int K2, int N) {
    int i = blockIdx.x * 256 + threadIdx.x;   // over (K/2)*N
    if (i < K2 * N) {
        int k2 = i / N, n = i - k2 * N;
        dst[i] = pack2bf(src[(size_t)(2 * k2) * N + n],
                         src[(size_t)(2 * k2 + 1) * N + n]);
    }
}

__global__ void zero_i32(int* p, int n) {
    int i = blockIdx.x * 32 + threadIdx.x;
    if (i < n) p[i] = 0;
}

// ---------------------------------------------------------------------------
// LayerNorm: one block per token. Writes f32 + bf16 (row-major); optionally
// copies input (residual base) into copy_to.
// ---------------------------------------------------------------------------
template <bool COPY>
__global__ __launch_bounds__(256)
void ln_kernel(const float* __restrict__ x, const float* __restrict__ g,
               const float* __restrict__ bta, float* __restrict__ yf,
               unsigned short* __restrict__ yb, float* __restrict__ copy_to) {
    __shared__ float ws1[8], ws2[8], mv[2];
    int t = threadIdx.x;
    int tok = blockIdx.x;
    const float* xr = x + (size_t)tok * D_;
    float s = 0.f, s2 = 0.f;
    for (int d = t; d < D_; d += 256) {
        float v = xr[d];
        s += v; s2 += v * v;
        if (COPY) copy_to[(size_t)tok * D_ + d] = v;
    }
    for (int off = 16; off > 0; off >>= 1) {
        s  += __shfl_down(s,  off, 32);
        s2 += __shfl_down(s2, off, 32);
    }
    if ((t & 31) == 0) { ws1[t >> 5] = s; ws2[t >> 5] = s2; }
    __syncthreads();
    if (t == 0) {
        float a = 0.f, b2 = 0.f;
        for (int i = 0; i < 8; i++) { a += ws1[i]; b2 += ws2[i]; }
        float m = a / D_;
        float var = b2 / D_ - m * m;
        mv[0] = m; mv[1] = rsqrtf(var + 1e-5f);
    }
    __syncthreads();
    float m = mv[0], ri = mv[1];
    for (int d = t; d < D_; d += 256) {
        float v = (xr[d] - m) * ri * g[d] + bta[d];
        yf[(size_t)tok * D_ + d] = v;
        yb[(size_t)tok * D_ + d] = f2bf(v);
    }
}

// ---------------------------------------------------------------------------
// bf16 WMMA GEMM, one wave computes a 16(M) x 64(N) tile: A fragment reused
// across 4 WMMAs per K-step; B read as pair-packed dwords (one b32 per vgpr).
// OUTB: 0 = none, 1 = row-major bf16, 2 = pair-packed bf16 (for V).
// ---------------------------------------------------------------------------
template <bool RES, bool OUTF, int OUTB>
__global__ __launch_bounds__(32)
void gemm_bf16_wmma(const unsigned short* __restrict__ A,
                    const unsigned int* __restrict__ Bp,
                    const float* __restrict__ residual,
                    float* __restrict__ Cf,
                    unsigned short* __restrict__ Cb,
                    unsigned int* __restrict__ Cp,
                    int M, int N, int Kd) {
    int l   = threadIdx.x;
    int col = l & 15;
    int hi  = (l >> 4) & 1;
    int n0  = blockIdx.x * 64;
    int m0  = blockIdx.y * 16;

    FragC c[4];
#pragma unroll
    for (int q = 0; q < 4; q++)
#pragma unroll
        for (int i = 0; i < 8; i++) c[q].f[i] = 0.f;

    const unsigned short* arp = A + (size_t)(m0 + col) * Kd;
    for (int k0 = 0; k0 < Kd; k0 += 32) {
        FragAB a;
#pragma unroll
        for (int v = 0; v < 8; v++)
            a.u[v] = *(const unsigned int*)(arp + k0 + akk(v, hi));
        const unsigned int* bpr = Bp + (size_t)((k0 >> 1) + (hi << 3)) * N + n0 + col;
#pragma unroll
        for (int q = 0; q < 4; q++) {
            FragAB b;
#pragma unroll
            for (int j = 0; j < 8; j++)
                b.u[j] = bpr[(size_t)j * N + q * 16];
            wmma_bf16(c[q], a, b);
        }
    }

#pragma unroll
    for (int q = 0; q < 4; q++) {
        int nq = n0 + q * 16 + col;
        float vals[8];
#pragma unroll
        for (int r = 0; r < 8; r++) {
            size_t idx = (size_t)(m0 + r + (hi << 3)) * N + nq;
            vals[r] = c[q].f[r];
            if (RES)  vals[r] += residual[idx];
            if (OUTF) Cf[idx] = vals[r];
            if (OUTB == 1) Cb[idx] = f2bf(vals[r]);
        }
        if (OUTB == 2) {
#pragma unroll
            for (int i = 0; i < 4; i++) {
                size_t idx = (size_t)((m0 >> 1) + (hi << 2) + i) * N + nq;
                Cp[idx] = pack2bf(vals[2 * i], vals[2 * i + 1]);
            }
        }
    }
}

// ---------------------------------------------------------------------------
// Causal attention, streaming softmax. One wave per (b, h, 16 query rows).
// S^T = K_tile(16x64) x Q^T(64x16); P pair-packed bf16 in LDS;
// O += P(16x32) x V(32x16) with V pair-packed (1 dword load per B vgpr).
// ---------------------------------------------------------------------------
__global__ __launch_bounds__(32)
void attn_kernel(const unsigned short* __restrict__ Qb,
                 const unsigned short* __restrict__ Kb,
                 const unsigned int*  __restrict__ Vp,
                 unsigned short* __restrict__ Ob) {
    __shared__ float St[32][16];
    __shared__ unsigned int Pp[16][16];   // pair-packed P: Pp[m/2][n]
    __shared__ float alpha_s[16], lsum_s[16];

    int l   = threadIdx.x;
    int col = l & 15;
    int hi  = (l >> 4) & 1;
    int qts = blockIdx.x * 16;
    int bh  = blockIdx.y;
    int b   = bh >> 4;
    int h   = bh & 15;
    int hoff = h * DH_;

    FragAB bq[2];
    const unsigned short* qrp = Qb + (size_t)(b * S_ + qts + col) * D_ + hoff;
#pragma unroll
    for (int half = 0; half < 2; half++)
#pragma unroll
        for (int j = 0; j < 8; j++)
            bq[half].u[j] = *(const unsigned int*)(qrp + half * 32 + (hi << 4) + (j << 1));

    FragC oc[4];
#pragma unroll
    for (int c4 = 0; c4 < 4; c4++)
#pragma unroll
        for (int r = 0; r < 8; r++) oc[c4].f[r] = 0.f;

    float mrow = -3.0e38f, lrow = 0.f;
    int nkb = (qts + 16 + 31) >> 5;

    for (int kb = 0; kb < nkb; kb++) {
#pragma unroll
        for (int sub = 0; sub < 2; sub++) {
            int krow = kb * 32 + sub * 16 + col;
            const unsigned short* krp = Kb + (size_t)(b * S_ + krow) * D_ + hoff;
            FragC cc;
#pragma unroll
            for (int i = 0; i < 8; i++) cc.f[i] = 0.f;
#pragma unroll
            for (int half = 0; half < 2; half++) {
                FragAB a;
#pragma unroll
                for (int v = 0; v < 8; v++)
                    a.u[v] = *(const unsigned int*)(krp + half * 32 + akk(v, hi));
                wmma_bf16(cc, a, bq[half]);
            }
#pragma unroll
            for (int r = 0; r < 8; r++)
                St[sub * 16 + r + (hi << 3)][col] = cc.f[r] * 0.125f;
        }
        __syncthreads();

        if (l < 16) {
            int qglob = qts + l;
            float sv[32], tm = -3.0e38f;
#pragma unroll
            for (int m = 0; m < 32; m++) {
                int key = kb * 32 + m;
                float s = (key <= qglob) ? St[m][l] : -3.0e38f;
                sv[m] = s;
                tm = fmaxf(tm, s);
            }
            float nm = fmaxf(mrow, tm);
            float al = __expf(mrow - nm);
            float ls = lrow * al;
            float pv[32];
#pragma unroll
            for (int m = 0; m < 32; m++) { pv[m] = __expf(sv[m] - nm); ls += pv[m]; }
#pragma unroll
            for (int m2 = 0; m2 < 16; m2++)
                Pp[m2][l] = pack2bf(pv[2 * m2], pv[2 * m2 + 1]);
            mrow = nm; lrow = ls;
            alpha_s[l] = al;
        }
        __syncthreads();

#pragma unroll
        for (int c4 = 0; c4 < 4; c4++)
#pragma unroll
            for (int r = 0; r < 8; r++)
                oc[c4].f[r] *= alpha_s[r + (hi << 3)];

        FragAB pa;
#pragma unroll
        for (int v = 0; v < 8; v++)
            pa.u[v] = Pp[hi * 8 + v][col];

#pragma unroll
        for (int c4 = 0; c4 < 4; c4++) {
            FragAB vb;
#pragma unroll
            for (int j = 0; j < 8; j++)
                vb.u[j] = Vp[(size_t)((b * S_ >> 1) + kb * 16 + (hi << 3) + j) * D_
                             + hoff + c4 * 16 + col];
            wmma_bf16(oc[c4], pa, vb);
        }
        __syncthreads();
    }

    if (l < 16) lsum_s[l] = 1.0f / lrow;
    __syncthreads();
#pragma unroll
    for (int c4 = 0; c4 < 4; c4++)
#pragma unroll
        for (int r = 0; r < 8; r++) {
            int m = r + (hi << 3);
            float val = oc[c4].f[r] * lsum_s[m];
            Ob[(size_t)(b * S_ + qts + m) * D_ + hoff + c4 * 16 + col] = f2bf(val);
        }
}

// ---------------------------------------------------------------------------
// Gate: one wave per token; top-2 softmax; compact token lists per expert.
// ---------------------------------------------------------------------------
__global__ __launch_bounds__(256)
void gate_kernel(const float* __restrict__ h2, const float* __restrict__ gw,
                 int* __restrict__ cnt, int* __restrict__ tokidx,
                 float* __restrict__ tokw) {
    int t = threadIdx.x;
    int wv = t >> 5;
    int l = t & 31;
    int tok = blockIdx.x * 8 + wv;
    const float* xr = h2 + (size_t)tok * D_;
    float acc[E_];
#pragma unroll
    for (int e = 0; e < E_; e++) acc[e] = 0.f;
    for (int d = l; d < D_; d += 32) {
        float xv = xr[d];
        const float* gr = gw + (size_t)d * E_;
#pragma unroll
        for (int e = 0; e < E_; e++) acc[e] += xv * gr[e];
    }
    for (int off = 16; off > 0; off >>= 1)
#pragma unroll
        for (int e = 0; e < E_; e++) acc[e] += __shfl_down(acc[e], off, 32);
    if (l == 0) {
        int i0 = 0;
#pragma unroll
        for (int e = 1; e < E_; e++) if (acc[e] > acc[i0]) i0 = e;
        int i1 = (i0 == 0) ? 1 : 0;
#pragma unroll
        for (int e = 0; e < E_; e++)
            if (e != i0 && acc[e] > acc[i1]) i1 = e;
        float e1 = __expf(acc[i1] - acc[i0]);
        float inv = 1.0f / (1.0f + e1);
        int s0 = atomicAdd(&cnt[i0], 1);
        tokidx[i0 * NTOK + s0] = tok; tokw[i0 * NTOK + s0] = inv;
        int s1 = atomicAdd(&cnt[i1], 1);
        tokidx[i1 * NTOK + s1] = tok; tokw[i1 * NTOK + s1] = e1 * inv;
    }
}

// ---------------------------------------------------------------------------
// Fused MoE FFN: grid (tiles, E). 8 waves / 16 gathered tokens / expert.
// A tile fetched via gfx1250 async global->LDS b128 copies (ASYNCcnt).
// Phase 1: H = gelu(A*W1+b1) -> LDS bf16. Phase 2: Y = H*W2 scaled+atomics.
// ---------------------------------------------------------------------------
__global__ __launch_bounds__(256)
void moe_kernel(const unsigned short* __restrict__ h2b,
                const unsigned int* __restrict__ w1p,
                const float* __restrict__ b1,
                const unsigned int* __restrict__ w2p,
                const float* __restrict__ b2,
                const int* __restrict__ cnt,
                const int* __restrict__ tokidx,
                const float* __restrict__ tokw,
                float* __restrict__ out) {
    __shared__ unsigned short Als[16][D_];   // 32 KB
    __shared__ unsigned short Hls[16][F_];   // 128 KB
    __shared__ int tks[16];
    __shared__ float wts[16];

    int e = blockIdx.y;
    int tile = blockIdx.x;
    int c = cnt[e];
    if (tile * 16 >= c) return;

    int t = threadIdx.x;
    if (t < 16) {
        int idx = tile * 16 + t;
        if (idx < c) { tks[t] = tokidx[e * NTOK + idx]; wts[t] = tokw[e * NTOK + idx]; }
        else         { tks[t] = tokidx[e * NTOK + tile * 16]; wts[t] = 0.f; }
    }
    __syncthreads();

    // gather 16 token rows (2KB each) into LDS with async b128 copies
    {
        unsigned ldsbase = (unsigned)(uintptr_t)&Als[0][0];
        for (int i = t; i < 16 * 128; i += 256) {           // 128 x 16B per row
            int r = i >> 7, cc = i & 127;
            const unsigned short* gp = h2b + (size_t)tks[r] * D_ + cc * 8;
            unsigned loff = ldsbase + r * (D_ * 2) + cc * 16;
            asm volatile("global_load_async_to_lds_b128 %0, %1, off"
                         :: "v"(loff), "v"(gp) : "memory");
        }
        asm volatile("s_wait_asynccnt 0" ::: "memory");
    }
    __syncthreads();

    int wv = t >> 5;
    int l = t & 31;
    int col = l & 15;
    int hi = (l >> 4) & 1;

    // phase 1: H = gelu(A * W1 + b1); each wave does 32 of 256 n-chunks
    for (int ncb = 0; ncb < 32; ncb++) {
        int n0 = (wv * 32 + ncb) * 16;
        FragC cc;
#pragma unroll
        for (int i = 0; i < 8; i++) cc.f[i] = 0.f;
        for (int k0 = 0; k0 < D_; k0 += 32) {
            FragAB a, bb;
#pragma unroll
            for (int v = 0; v < 8; v++)
                a.u[v] = *(const unsigned int*)&Als[col][k0 + akk(v, hi)];
#pragma unroll
            for (int j = 0; j < 8; j++)
                bb.u[j] = w1p[(size_t)(((e * D_ + k0) >> 1) + (hi << 3) + j) * F_ + n0 + col];
            wmma_bf16(cc, a, bb);
        }
#pragma unroll
        for (int r = 0; r < 8; r++) {
            int m = r + (hi << 3);
            float xv = cc.f[r] + b1[e * F_ + n0 + col];
            Hls[m][n0 + col] = f2bf(gelu_tanh(xv));
        }
    }
    __syncthreads();

    // phase 2: Y = H * W2; each wave does 8 of 64 n-chunks
    for (int yc = 0; yc < 8; yc++) {
        int n0 = (wv * 8 + yc) * 16;
        FragC cc;
#pragma unroll
        for (int i = 0; i < 8; i++) cc.f[i] = 0.f;
        for (int k0 = 0; k0 < F_; k0 += 32) {
            FragAB a, bb;
#pragma unroll
            for (int v = 0; v < 8; v++)
                a.u[v] = *(const unsigned int*)&Hls[col][k0 + akk(v, hi)];
#pragma unroll
            for (int j = 0; j < 8; j++)
                bb.u[j] = w2p[(size_t)(((e * F_ + k0) >> 1) + (hi << 3) + j) * D_ + n0 + col];
            wmma_bf16(cc, a, bb);
        }
#pragma unroll
        for (int r = 0; r < 8; r++) {
            int m = r + (hi << 3);
            float val = (cc.f[r] + b2[e * D_ + n0 + col]) * wts[m];
            atomicAdd(&out[(size_t)tks[m] * D_ + n0 + col], val);
        }
    }
}

// ---------------------------------------------------------------------------
// Host-side launch
// ---------------------------------------------------------------------------
extern "C" void kernel_launch(void* const* d_in, const int* in_sizes, int n_in,
                              void* d_out, int out_size, void* d_ws, size_t ws_size,
                              hipStream_t stream) {
    const float* x      = (const float*)d_in[0];
    const float* wq     = (const float*)d_in[2];
    const float* wk     = (const float*)d_in[3];
    const float* wv     = (const float*)d_in[4];
    const float* wo     = (const float*)d_in[5];
    const float* ln1_g  = (const float*)d_in[6];
    const float* ln1_b  = (const float*)d_in[7];
    const float* ln2_g  = (const float*)d_in[8];
    const float* ln2_b  = (const float*)d_in[9];
    const float* gate_w = (const float*)d_in[10];
    const float* e_w1   = (const float*)d_in[11];
    const float* e_b1   = (const float*)d_in[12];
    const float* e_w2   = (const float*)d_in[13];
    const float* e_b2   = (const float*)d_in[14];
    float* out = (float*)d_out;

    char* w = (char*)d_ws;
    auto alloc = [&](size_t bytes) -> void* {
        void* p = (void*)w;
        w += (bytes + 255) & ~(size_t)255;
        return p;
    };
    const size_t ND = (size_t)NTOK * D_;
    float*          h1  = (float*)alloc(ND * 4);
    unsigned short* h1b = (unsigned short*)alloc(ND * 2);
    unsigned int*   wqp = (unsigned int*)alloc((size_t)D_ * D_ * 2);
    unsigned int*   wkp = (unsigned int*)alloc((size_t)D_ * D_ * 2);
    unsigned int*   wvp = (unsigned int*)alloc((size_t)D_ * D_ * 2);
    unsigned int*   wop = (unsigned int*)alloc((size_t)D_ * D_ * 2);
    unsigned short* Qb  = (unsigned short*)alloc(ND * 2);
    unsigned short* Kb  = (unsigned short*)alloc(ND * 2);
    unsigned int*   Vp  = (unsigned int*)alloc(ND * 2);
    unsigned short* Ob  = (unsigned short*)alloc(ND * 2);
    float*          x2  = (float*)alloc(ND * 4);
    float*          h2  = (float*)alloc(ND * 4);
    unsigned short* h2b = (unsigned short*)alloc(ND * 2);
    int*            cnt = (int*)alloc(E_ * 4);
    int*            tki = (int*)alloc((size_t)E_ * NTOK * 4);
    float*          tkw = (float*)alloc((size_t)E_ * NTOK * 4);
    unsigned int*   w1p = (unsigned int*)alloc((size_t)E_ * D_ * F_ * 2);
    unsigned int*   w2p = (unsigned int*)alloc((size_t)E_ * D_ * F_ * 2);

    auto cvtp = [&](const float* s, unsigned int* d, int K, int N) {
        size_t n = (size_t)(K / 2) * N;
        cvt_pack_rows<<<dim3((unsigned)((n + 255) / 256)), dim3(256), 0, stream>>>(
            s, d, K / 2, N);
    };
    cvtp(wq, wqp, D_, D_);
    cvtp(wk, wkp, D_, D_);
    cvtp(wv, wvp, D_, D_);
    cvtp(wo, wop, D_, D_);
    cvtp(e_w1, w1p, E_ * D_, F_);
    cvtp(e_w2, w2p, E_ * F_, D_);

    // LN1
    ln_kernel<false><<<dim3(NTOK), dim3(256), 0, stream>>>(x, ln1_g, ln1_b, h1, h1b, nullptr);

    // Q, K (row-major bf16), V (pair-packed bf16)
    dim3 ggrid(D_ / 64, NTOK / 16);
    gemm_bf16_wmma<false, false, 1><<<ggrid, dim3(32), 0, stream>>>(
        h1b, wqp, nullptr, nullptr, Qb, nullptr, NTOK, D_, D_);
    gemm_bf16_wmma<false, false, 1><<<ggrid, dim3(32), 0, stream>>>(
        h1b, wkp, nullptr, nullptr, Kb, nullptr, NTOK, D_, D_);
    gemm_bf16_wmma<false, false, 2><<<ggrid, dim3(32), 0, stream>>>(
        h1b, wvp, nullptr, nullptr, nullptr, Vp, NTOK, D_, D_);

    // attention
    attn_kernel<<<dim3(S_ / 16, B_ * H_), dim3(32), 0, stream>>>(Qb, Kb, Vp, Ob);

    // output projection + residual (x2 = h1 + O @ wo)
    gemm_bf16_wmma<true, true, 0><<<ggrid, dim3(32), 0, stream>>>(
        Ob, wop, h1, x2, nullptr, nullptr, NTOK, D_, D_);

    // LN2 (also seeds d_out with residual base x2)
    ln_kernel<true><<<dim3(NTOK), dim3(256), 0, stream>>>(x2, ln2_g, ln2_b, h2, h2b, out);

    // gating
    zero_i32<<<dim3(1), dim3(32), 0, stream>>>(cnt, E_);
    gate_kernel<<<dim3(NTOK / 8), dim3(256), 0, stream>>>(h2, gate_w, cnt, tki, tkw);

    // fused expert FFN, atomically accumulates into out
    moe_kernel<<<dim3(NTOK / 16, E_), dim3(256), 0, stream>>>(h2b, w1p, e_b1, w2p, e_b2,
                                                              cnt, tki, tkw, out);
}
